// SpanClassModel_79843442032823
// MI455X (gfx1250) — matrix operand order repaired and checked by reading, not compiled, not run on hardware
//
#include <hip/hip_runtime.h>
#include <hip/hip_bf16.h>
#include <math.h>

// Problem constants (from reference)
#define BB 16
#define SS 512
#define DD 768
#define LL 5
#define MAXW 5
#define TOPK 8
#define NSPAN (SS * MAXW)          // 2560 spans per batch
#define KCHUNKS 24                 // 768 / 32

typedef __bf16 bf16_t;
typedef bf16_t v16bf __attribute__((ext_vector_type(16)));
typedef float  v8f   __attribute__((ext_vector_type(8)));
typedef unsigned int u32x4 __attribute__((ext_vector_type(4)));
typedef int          i32x4 __attribute__((ext_vector_type(4)));
typedef int          i32x8 __attribute__((ext_vector_type(8)));

union Frag { uint4 q[2]; v16bf v; };

#if __has_builtin(__builtin_amdgcn_tensor_load_to_lds) && __has_builtin(__builtin_amdgcn_s_wait_tensorcnt)
#define USE_TDM 1
#endif

// float -> bf16 round-to-nearest-even, and back
static __device__ __forceinline__ unsigned short f2bf(float x) {
    union { float f; unsigned u; } v; v.f = x;
    const unsigned r = v.u + 0x7fffu + ((v.u >> 16) & 1u);
    return (unsigned short)(r >> 16);
}
static __device__ __forceinline__ float bf2f(unsigned short h) {
    union { unsigned u; float f; } v; v.u = ((unsigned)h) << 16;
    return v.f;
}

#ifdef USE_TDM
static __device__ __forceinline__ unsigned lds_lo32(const void* p) {
    // LDS aperture: flat addr[31:0] is the LDS address (ISA 10.2)
    return (unsigned)(unsigned long long)p;
}
// 2D tile load via Tensor Data Mover. Pads LDS dest: +4 dwords per 16 dwords
// (pad_interval code 3 = 16 DW, pad_amount code 3 = 4 DW) -> 80B LDS rows.
static __device__ __forceinline__ void tdm_load_tile(
    const void* gaddr, unsigned lds_addr, unsigned data_size_code,
    unsigned tensor_d0, unsigned tensor_d1, unsigned stride0,
    unsigned tile_d0, unsigned tile_d1)
{
    const unsigned long long ga = (unsigned long long)gaddr;
    u32x4 g0;
    g0.x = 1u;                                                  // count=1 (valid user D#)
    g0.y = lds_addr;                                            // lds_addr [63:32]
    g0.z = (unsigned)(ga & 0xffffffffu);                        // global_addr lo
    g0.w = (unsigned)((ga >> 32) & 0x1ffffffu) | (2u << 30);    // addr[56:32] | type=2
    i32x8 g1;
    g1[0] = (int)((data_size_code << 16) | (1u << 20) | (3u << 22) | (3u << 25));
    g1[1] = (int)((tensor_d0 & 0xffffu) << 16);                 // abar=0 | tensor_dim0 lo16
    g1[2] = (int)((tensor_d0 >> 16) | ((tensor_d1 & 0xffffu) << 16));
    g1[3] = (int)((tensor_d1 >> 16) | (tile_d0 << 16));         // tensor_dim1 hi | tile_dim0
    g1[4] = (int)(tile_d1 & 0xffffu);                           // tile_dim1 | tile_dim2=0
    g1[5] = (int)stride0;                                       // tensor_dim0_stride lo32
    g1[6] = 0;                                                  // stride0 hi | stride1 lo
    g1[7] = 0;                                                  // stride1 hi
    const i32x4 z4 = {0, 0, 0, 0};
#if __clang_major__ >= 23
    const i32x8 z8 = {0, 0, 0, 0, 0, 0, 0, 0};
    __builtin_amdgcn_tensor_load_to_lds(g0, g1, z4, z4, z8, 0);
#else
    __builtin_amdgcn_tensor_load_to_lds(g0, g1, z4, z4, 0);
#endif
}
#endif // USE_TDM

// ---------------------------------------------------------------------------
// Prep A: token_embs (8192x768 f32) -> Ahi/Alo bf16 row-major (hi+lo split)
// ---------------------------------------------------------------------------
__global__ __launch_bounds__(256)
void prep_a_kernel(const float* __restrict__ A,
                   unsigned short* __restrict__ hi, unsigned short* __restrict__ lo)
{
    const size_t i4 = ((size_t)blockIdx.x * 256 + threadIdx.x) * 4;
    const float4 v = *reinterpret_cast<const float4*>(A + i4);
    ushort4 h, l;
    h.x = f2bf(v.x); l.x = f2bf(v.x - bf2f(h.x));
    h.y = f2bf(v.y); l.y = f2bf(v.y - bf2f(h.y));
    h.z = f2bf(v.z); l.z = f2bf(v.z - bf2f(h.z));
    h.w = f2bf(v.w); l.w = f2bf(v.w - bf2f(h.w));
    *reinterpret_cast<ushort4*>(hi + i4) = h;
    *reinterpret_cast<ushort4*>(lo + i4) = l;
}

// ---------------------------------------------------------------------------
// Prep W: W1 (1536x768 f32) -> hi/lo packed K-pair dwords in WMMA-B order:
//   dword[((half*24 + c)*768 + n)*16 + k2] = { W[hc*32+2k2][n], W[hc*32+2k2+1][n] }
// ---------------------------------------------------------------------------
__global__ __launch_bounds__(256)
void prep_w_kernel(const float* __restrict__ W1,
                   unsigned int* __restrict__ whi, unsigned int* __restrict__ wlo)
{
    const int g  = blockIdx.x * 256 + threadIdx.x;   // ((hc)*16 + k2)*768 + n
    const int n  = g % DD;
    const int r  = g / DD;
    const int k2 = r % 16;
    const int hc = r / 16;                           // half*24 + c  (0..47)
    const int krow = hc * 32 + 2 * k2;               // == half*768 + c*32 + 2k2
    const float a0 = W1[(size_t)krow * DD + n];
    const float a1 = W1[(size_t)(krow + 1) * DD + n];
    const unsigned short h0 = f2bf(a0), h1 = f2bf(a1);
    const unsigned short l0 = f2bf(a0 - bf2f(h0)), l1 = f2bf(a1 - bf2f(h1));
    const size_t o = ((size_t)hc * DD + n) * 16 + k2;
    whi[o] = (unsigned)h0 | ((unsigned)h1 << 16);
    wlo[o] = (unsigned)l0 | ((unsigned)l1 << 16);
}

// ---------------------------------------------------------------------------
// Kernel 1: P = tokens @ W1[0:768,:], Q = tokens @ W1[768:,:]
// Split-precision bf16x3 WMMA (16x16x32), fp32 accumulate.
// TDM double-buffered LDS staging (fallback: sync copies).
// ---------------------------------------------------------------------------
#define TILE_M 128
#define TILE_N 64
#define CHUNK_K 32
#define LDA_H 40   // bf16 per A row in LDS (32 data + 8 pad) -> 16+4 dwords
#define LDB_W 20   // dwords per B row in LDS (16 data + 4 pad)

__global__ __launch_bounds__(256)
void gemm_pq_bf16x3_kernel(const unsigned short* __restrict__ Ahi,
                           const unsigned short* __restrict__ Alo,
                           const unsigned int* __restrict__ Whi,
                           const unsigned int* __restrict__ Wlo,
                           float* __restrict__ P, float* __restrict__ Q)
{
#ifdef USE_TDM
    __shared__ unsigned short sAhi[2][TILE_M * LDA_H];
    __shared__ unsigned short sAlo[2][TILE_M * LDA_H];
    __shared__ unsigned int   sBhi[2][TILE_N * LDB_W];
    __shared__ unsigned int   sBlo[2][TILE_N * LDB_W];
#else
    __shared__ unsigned short sAhi[1][TILE_M * LDA_H];
    __shared__ unsigned short sAlo[1][TILE_M * LDA_H];
    __shared__ unsigned int   sBhi[1][TILE_N * LDB_W];
    __shared__ unsigned int   sBlo[1][TILE_N * LDB_W];
#endif

    const int half = blockIdx.z;                     // 0 -> P, 1 -> Q
    float* __restrict__ OUT = half ? Q : P;

    const int tile_n = blockIdx.x;
    const int tile_m = blockIdx.y;
    const int tid  = threadIdx.x;
    const int wave = tid >> 5;
    const int lane = tid & 31;
    const int l16  = lane & 15;
    const int hi   = lane >> 4;

    const int m_base = tile_m * TILE_M;
    const int n_base = tile_n * TILE_N;

    v8f acc[4] = {};

#ifdef USE_TDM
    // ---- issue helper: 4 TDM descriptors fill one buffer with chunk c ----
    auto issue = [&](int c, int buf) {
        const size_t ga = (size_t)m_base * DD + c * CHUNK_K;           // bf16 elems
        const size_t gb = ((size_t)(half * KCHUNKS + c) * DD + n_base) * 16; // dwords
        // A: 2D tile, 128 rows x 32 bf16, row stride 768 elems
        tdm_load_tile(Ahi + ga, lds_lo32(&sAhi[buf][0]), 1u,
                      DD, (unsigned)(BB * SS), DD, CHUNK_K, TILE_M);
        tdm_load_tile(Alo + ga, lds_lo32(&sAlo[buf][0]), 1u,
                      DD, (unsigned)(BB * SS), DD, CHUNK_K, TILE_M);
        // B: contiguous 1D stream of 1024 dwords (packed layout)
        tdm_load_tile(Whi + gb, lds_lo32(&sBhi[buf][0]), 2u,
                      (unsigned)(2 * KCHUNKS * DD * 16), 1u, 1024u, 1024u, 1u);
        tdm_load_tile(Wlo + gb, lds_lo32(&sBlo[buf][0]), 2u,
                      (unsigned)(2 * KCHUNKS * DD * 16), 1u, 1024u, 1024u, 1u);
    };
    if (wave == 0) issue(0, 0);
#endif

    for (int c = 0; c < KCHUNKS; ++c) {
#ifdef USE_TDM
        const int buf = c & 1;
        if (wave == 0) {
            if (c + 1 < KCHUNKS) {
                issue(c + 1, (c + 1) & 1);               // overlap next-chunk DMA
                __builtin_amdgcn_s_wait_tensorcnt((short)1); // chunk c landed
            } else {
                __builtin_amdgcn_s_wait_tensorcnt((short)0);
            }
        }
        __syncthreads();
#else
        const int buf = 0;
        const int k0 = c * CHUNK_K;
        for (int i = tid; i < TILE_M * 4; i += 256) {
            const int row = i >> 2;
            const int j   = (i & 3) * 8;                 // 8 bf16 = 16 B
            const size_t ga = (size_t)(m_base + row) * DD + k0 + j;
            *reinterpret_cast<uint4*>(&sAhi[0][row * LDA_H + j]) =
                *reinterpret_cast<const uint4*>(&Ahi[ga]);
            *reinterpret_cast<uint4*>(&sAlo[0][row * LDA_H + j]) =
                *reinterpret_cast<const uint4*>(&Alo[ga]);
        }
        {
            const int n  = tid >> 2;
            const int k4 = (tid & 3) * 4;
            const size_t gb = ((size_t)(half * KCHUNKS + c) * DD + n_base + n) * 16 + k4;
            *reinterpret_cast<uint4*>(&sBhi[0][n * LDB_W + k4]) =
                *reinterpret_cast<const uint4*>(&Whi[gb]);
            *reinterpret_cast<uint4*>(&sBlo[0][n * LDB_W + k4]) =
                *reinterpret_cast<const uint4*>(&Wlo[gb]);
        }
        if (c + 1 < KCHUNKS) {
            __builtin_prefetch(&Ahi[(size_t)(m_base + (tid >> 1)) * DD + k0 + CHUNK_K], 0, 1);
            __builtin_prefetch(&Whi[((size_t)(half * KCHUNKS + c + 1) * DD + n_base + (tid & 63)) * 16], 0, 1);
        }
        __syncthreads();
#endif

        // ---- A fragments (16-bit A 16x32 layout): two b128 each ----
        const int am = wave * 16 + l16;
        Frag ah, al;
        ah.q[0] = *reinterpret_cast<const uint4*>(&sAhi[buf][am * LDA_H + hi * 8]);
        ah.q[1] = *reinterpret_cast<const uint4*>(&sAhi[buf][am * LDA_H + 16 + hi * 8]);
        al.q[0] = *reinterpret_cast<const uint4*>(&sAlo[buf][am * LDA_H + hi * 8]);
        al.q[1] = *reinterpret_cast<const uint4*>(&sAlo[buf][am * LDA_H + 16 + hi * 8]);

        // ---- 4 N-tiles x 3 split products = 12 WMMA per chunk per wave ----
        #pragma unroll
        for (int nt = 0; nt < 4; ++nt) {
            const int bn = nt * 16 + l16;
            Frag bh, bl;
            bh.q[0] = *reinterpret_cast<const uint4*>(&sBhi[buf][bn * LDB_W + hi * 4]);
            bh.q[1] = *reinterpret_cast<const uint4*>(&sBhi[buf][bn * LDB_W + 8 + hi * 4]);
            bl.q[0] = *reinterpret_cast<const uint4*>(&sBlo[buf][bn * LDB_W + hi * 4]);
            bl.q[1] = *reinterpret_cast<const uint4*>(&sBlo[buf][bn * LDB_W + 8 + hi * 4]);
            acc[nt] = __builtin_amdgcn_wmma_f32_16x16x32_bf16(
                false, ah.v, false, bh.v, (short)0, acc[nt], false, false);
            acc[nt] = __builtin_amdgcn_wmma_f32_16x16x32_bf16(
                false, ah.v, false, bl.v, (short)0, acc[nt], false, false);
            acc[nt] = __builtin_amdgcn_wmma_f32_16x16x32_bf16(
                false, al.v, false, bh.v, (short)0, acc[nt], false, false);
        }
        __syncthreads();
    }

    // ---- store: each wave writes its 16x64 strip ----
    #pragma unroll
    for (int nt = 0; nt < 4; ++nt) {
        const int col = n_base + nt * 16 + l16;
        #pragma unroll
        for (int r = 0; r < 8; ++r) {
            const int row = m_base + wave * 16 + hi * 8 + r;
            OUT[(size_t)row * DD + col] = acc[nt][r];
        }
    }
}

// ---------------------------------------------------------------------------
// Kernel 2a: v[d] = sum_e W2[d,e] * Ws[e]
// ---------------------------------------------------------------------------
__global__ __launch_bounds__(256)
void wv_kernel(const float* __restrict__ W2, const float* __restrict__ Ws,
               float* __restrict__ v)
{
    const int d = blockIdx.x * 256 + threadIdx.x;
    if (d < DD) {
        float acc = 0.f;
        const float* row = W2 + (size_t)d * DD;
        for (int e = 0; e < DD; ++e) acc += row[e] * Ws[e];
        v[d] = acc;
    }
}

// Kernel 2b: c = dot(b2, Ws) + bs
__global__ void c_kernel(const float* __restrict__ b2, const float* __restrict__ Ws,
                         const float* __restrict__ bs, float* __restrict__ cptr)
{
    const int lane = threadIdx.x & 31;
    float acc = 0.f;
    for (int e = lane; e < DD; e += 32) acc += b2[e] * Ws[e];
    for (int off = 16; off > 0; off >>= 1) acc += __shfl_xor(acc, off, 32);
    if (lane == 0) cptr[0] = acc + bs[0];
}

// Kernel 2c: seq_len[b] = sum(token_mask[b,:])
__global__ __launch_bounds__(256)
void seqlen_kernel(const int* __restrict__ mask, int* __restrict__ seq_len)
{
    __shared__ int red[256];
    const int b = blockIdx.x;
    const int t = threadIdx.x;
    int sum = 0;
    for (int i = t; i < SS; i += 256) sum += mask[b * SS + i];
    red[t] = sum;
    __syncthreads();
    for (int s = 128; s > 0; s >>= 1) {
        if (t < s) red[t] += red[t + s];
        __syncthreads();
    }
    if (t == 0) seq_len[b] = red[0];
}

// ---------------------------------------------------------------------------
// Kernel 3: masked span scores. One wave per span.
// ---------------------------------------------------------------------------
__global__ __launch_bounds__(256)
void span_score_kernel(const float* __restrict__ P, const float* __restrict__ Q,
                       const float* __restrict__ b1, const float* __restrict__ v,
                       const float* __restrict__ cptr, const int* __restrict__ seq_len,
                       float* __restrict__ scores)
{
    const int wave = threadIdx.x >> 5;
    const int lane = threadIdx.x & 31;
    const int span = blockIdx.x * 8 + wave;
    const int b = span / NSPAN;
    const int n = span % NSPAN;
    const int s = n / MAXW;
    const int w = n % MAXW;
    const int e = min(s + w, SS - 1);

    const float* pr = P + ((size_t)(b * SS + s)) * DD;
    const float* qr = Q + ((size_t)(b * SS + e)) * DD;

    float acc = 0.f;
    for (int d = lane; d < DD; d += 32) {
        float h = pr[d] + qr[d] + b1[d];
        h = fmaxf(h, 0.f);
        acc += h * v[d];
    }
    for (int off = 16; off > 0; off >>= 1) acc += __shfl_xor(acc, off, 32);
    if (lane == 0) {
        const bool valid = (s + w) < seq_len[b];
        scores[span] = valid ? (acc + cptr[0]) : -1e30f;
    }
}

// ---------------------------------------------------------------------------
// Kernel 4: top-8 per batch (iterative argmax, smaller-index tiebreak)
// ---------------------------------------------------------------------------
__global__ __launch_bounds__(256)
void topk_kernel(const float* __restrict__ scores,
                 float* __restrict__ top_rel, int* __restrict__ top_idx)
{
    __shared__ float sv[256];
    __shared__ int   si[256];
    __shared__ int   sel[TOPK];
    const int b = blockIdx.x;
    const int t = threadIdx.x;
    const float* sc = scores + (size_t)b * NSPAN;

    for (int k = 0; k < TOPK; ++k) {
        float best = -3.0e38f;
        int bidx = 0x7fffffff;
        for (int n = t; n < NSPAN; n += 256) {
            bool taken = false;
            for (int j = 0; j < k; ++j) taken = taken || (sel[j] == n);
            if (taken) continue;
            const float val = sc[n];
            if (val > best || (val == best && n < bidx)) { best = val; bidx = n; }
        }
        sv[t] = best; si[t] = bidx;
        __syncthreads();
        for (int s = 128; s > 0; s >>= 1) {
            if (t < s) {
                const float v2 = sv[t + s]; const int i2 = si[t + s];
                if (v2 > sv[t] || (v2 == sv[t] && i2 < si[t])) { sv[t] = v2; si[t] = i2; }
            }
            __syncthreads();
        }
        if (t == 0) {
            sel[k] = si[0];
            top_rel[b * TOPK + k] = sv[0];
            top_idx[b * TOPK + k] = si[0];
        }
        __syncthreads();
    }
}

// ---------------------------------------------------------------------------
// Kernel 5: finisher (softmax factors out of the label matmul)
// ---------------------------------------------------------------------------
__global__ __launch_bounds__(256)
void final_kernel(const float* __restrict__ P, const float* __restrict__ Q,
                  const float* __restrict__ b1, const float* __restrict__ W2,
                  const float* __restrict__ b2, const float* __restrict__ label_embs,
                  const float* __restrict__ top_rel, const int* __restrict__ top_idx,
                  float* __restrict__ out)
{
    __shared__ float attn[TOPK];
    __shared__ int   ss[TOPK], ee[TOPK];
    __shared__ float hbar[DD];
    __shared__ float u[DD];

    const int b = blockIdx.x;
    const int t = threadIdx.x;

    if (t < TOPK) {
        const int n = top_idx[b * TOPK + t];
        const int s = n / MAXW;
        const int w = n % MAXW;
        ss[t] = s;
        ee[t] = min(s + w, SS - 1);
    }
    if (t == 0) {
        float r[TOPK];
        float m = -3.0e38f;
        for (int k = 0; k < TOPK; ++k) { r[k] = top_rel[b * TOPK + k]; m = fmaxf(m, r[k]); }
        float sum = 0.f;
        for (int k = 0; k < TOPK; ++k) { r[k] = expf(r[k] - m); sum += r[k]; }
        for (int k = 0; k < TOPK; ++k) attn[k] = r[k] / sum;
    }
    __syncthreads();

    for (int d = t; d < DD; d += 256) {
        float acc = 0.f;
        const float bb = b1[d];
        #pragma unroll
        for (int k = 0; k < TOPK; ++k) {
            const float h = P[((size_t)(b * SS + ss[k])) * DD + d]
                          + Q[((size_t)(b * SS + ee[k])) * DD + d] + bb;
            acc += attn[k] * fmaxf(h, 0.f);
        }
        hbar[d] = acc;
    }
    __syncthreads();

    for (int e = t; e < DD; e += 256) {
        float acc = b2[e];
        for (int d = 0; d < DD; ++d) acc += hbar[d] * W2[(size_t)d * DD + e];
        u[e] = acc;
    }
    __syncthreads();

    const int wave = t >> 5, lane = t & 31;
    if (wave < LL) {
        const float* le = label_embs + ((size_t)b * LL + wave) * DD;
        float acc = 0.f;
        for (int e = lane; e < DD; e += 32) acc += le[e] * u[e];
        for (int off = 16; off > 0; off >>= 1) acc += __shfl_xor(acc, off, 32);
        if (lane == 0) {
            out[b * LL + wave] = 1.f / (1.f + expf(-acc));      // scores
            out[BB * LL + b * LL + wave] = 1.0f;                // mask (all true)
        }
    }
}

// ---------------------------------------------------------------------------
extern "C" void kernel_launch(void* const* d_in, const int* in_sizes, int n_in,
                              void* d_out, int out_size, void* d_ws, size_t ws_size,
                              hipStream_t stream)
{
    (void)in_sizes; (void)n_in; (void)out_size; (void)ws_size;

    const float* token_embs = (const float*)d_in[0];   // (16,512,768)
    const int*   token_mask = (const int*)  d_in[1];   // (16,512)
    const float* label_embs = (const float*)d_in[2];   // (16,5,768)
    const float* W1         = (const float*)d_in[3];   // (1536,768)
    const float* b1         = (const float*)d_in[4];   // (768)
    const float* W2         = (const float*)d_in[5];   // (768,768)
    const float* b2         = (const float*)d_in[6];   // (768)
    const float* Ws         = (const float*)d_in[7];   // (768)
    const float* bs         = (const float*)d_in[8];   // (1)
    float* out = (float*)d_out;

    // Workspace carve-up (256B aligned)
    char* ws = (char*)d_ws;
    size_t off = 0;
    auto carve = [&](size_t bytes) -> char* {
        char* p = ws + off;
        off = (off + bytes + 255) & ~(size_t)255;
        return p;
    };
    float*          P       = (float*)carve((size_t)BB * SS * DD * sizeof(float));
    float*          Q       = (float*)carve((size_t)BB * SS * DD * sizeof(float));
    unsigned short* Ahi     = (unsigned short*)carve((size_t)BB * SS * DD * 2);
    unsigned short* Alo     = (unsigned short*)carve((size_t)BB * SS * DD * 2);
    unsigned int*   Whi     = (unsigned int*)carve((size_t)2 * KCHUNKS * DD * 16 * 4);
    unsigned int*   Wlo     = (unsigned int*)carve((size_t)2 * KCHUNKS * DD * 16 * 4);
    float*          v       = (float*)carve(DD * sizeof(float));
    float*          cptr    = (float*)carve(16);
    int*            seqlen  = (int*)  carve(BB * sizeof(int));
    float*          scores  = (float*)carve((size_t)BB * NSPAN * sizeof(float));
    float*          top_rel = (float*)carve(BB * TOPK * sizeof(float));
    int*            top_idx = (int*)  carve(BB * TOPK * sizeof(int));

    // Precision split + layout swizzle (one-time)
    prep_a_kernel<<<dim3((BB * SS * DD) / 1024), dim3(256), 0, stream>>>(token_embs, Ahi, Alo);
    prep_w_kernel<<<dim3((2 * KCHUNKS * 16 * DD) / 256), dim3(256), 0, stream>>>(W1, Whi, Wlo);

    // Small folds
    wv_kernel<<<dim3((DD + 255) / 256), dim3(256), 0, stream>>>(W2, Ws, v);
    c_kernel<<<dim3(1), dim3(32), 0, stream>>>(b2, Ws, bs, cptr);
    seqlen_kernel<<<dim3(BB), dim3(256), 0, stream>>>(token_mask, seqlen);

    // Dominant GEMM: split-precision bf16x3 WMMA, TDM-staged double buffering
    gemm_pq_bf16x3_kernel<<<dim3(DD / TILE_N, (BB * SS) / TILE_M, 2), dim3(256), 0, stream>>>(
        Ahi, Alo, Whi, Wlo, P, Q);

    // Per-span scoring (one wave per span)
    span_score_kernel<<<dim3((BB * NSPAN) / 8), dim3(256), 0, stream>>>(
        P, Q, b1, v, cptr, seqlen, scores);

    // Top-8 per batch
    topk_kernel<<<dim3(BB), dim3(256), 0, stream>>>(scores, top_rel, top_idx);

    // Fused softmax/attention/label scoring + sigmoid
    final_kernel<<<dim3(BB), dim3(256), 0, stream>>>(
        P, Q, b1, W2, b2, label_embs, top_rel, top_idx, out);
}
